// HGT_15848429322698
// MI455X (gfx1250) — compile-verified, matrix operand order checked by reading, not run on hardware
//
#include <hip/hip_runtime.h>
#include <math.h>

typedef __attribute__((ext_vector_type(16))) _Float16 v16h;
typedef __attribute__((ext_vector_type(8)))  _Float16 v8h;
typedef __attribute__((ext_vector_type(8)))  float    v8f;
typedef __attribute__((ext_vector_type(4)))  unsigned u32x4;
typedef __attribute__((ext_vector_type(4)))  int      i32x4;
typedef __attribute__((ext_vector_type(8)))  int      i32x8;

#define CAT16(lo, hi) __builtin_shufflevector((lo), (hi), 0,1,2,3,4,5,6,7,8,9,10,11,12,13,14,15)

#if defined(__has_builtin)
#if __has_builtin(__builtin_amdgcn_tensor_load_to_lds) && __has_builtin(__builtin_amdgcn_s_wait_tensorcnt)
#define HGT_HAS_TDM 1
#endif
#endif

// ---------------------------------------------------------------------------
// helpers
// ---------------------------------------------------------------------------
__device__ __forceinline__ float gelu_exact(float x) {
    return 0.5f * x * (1.0f + erff(x * 0.70710678118654752f));
}
__device__ __forceinline__ unsigned enc_ord(float f) {
    unsigned u = __float_as_uint(f);
    return (u & 0x80000000u) ? ~u : (u | 0x80000000u);
}
__device__ __forceinline__ float dec_ord(unsigned u) {
    u = (u & 0x80000000u) ? (u & 0x7FFFFFFFu) : ~u;
    return __uint_as_float(u);
}
#define ENC_NEG_INF 0x007FFFFFu   // enc_ord(-inf)

// ---------------------------------------------------------------------------
// fills
// ---------------------------------------------------------------------------
__global__ void hgt_fill_f32(float* __restrict__ p, float v, long n) {
    long i = (long)blockIdx.x * blockDim.x + threadIdx.x;
    if (i < n) p[i] = v;
}
__global__ void hgt_fill_u32(unsigned* __restrict__ p, unsigned v, long n) {
    long i = (long)blockIdx.x * blockDim.x + threadIdx.x;
    if (i < n) p[i] = v;
}

// ---------------------------------------------------------------------------
// weight prep: f32 [K x 128] -> f16 transposed [128 x K] (n-major)
// ---------------------------------------------------------------------------
__global__ void hgt_prep_plain(const float* __restrict__ W, int K, _Float16* __restrict__ Wt) {
    int idx = blockIdx.x * 256 + threadIdx.x;       // K*128 total
    if (idx >= K * 128) return;
    int n = idx & 127;
    int k = idx >> 7;
    Wt[(size_t)n * K + k] = (_Float16)W[(size_t)k * 128 + n];
}

// fold per-head relation matrix into projection weights:
// Wt_eff[n=h*16+e][kin] = sum_d W[kin][h*16+d] * rel[h][d][e]; beff similarly
__global__ void hgt_prep_rel(const float* __restrict__ W, const float* __restrict__ b,
                             const float* __restrict__ rel,
                             _Float16* __restrict__ Wt, float* __restrict__ beff) {
    int idx = blockIdx.x * 256 + threadIdx.x;       // 128*128 total
    if (idx >= 128 * 128) return;
    int n   = idx & 127;
    int kin = idx >> 7;
    int h = n >> 4, e = n & 15;
    float s = 0.0f;
#pragma unroll
    for (int d = 0; d < 16; ++d)
        s = fmaf(W[kin * 128 + h * 16 + d], rel[(h * 16 + d) * 16 + e], s);
    Wt[(size_t)n * 128 + kin] = (_Float16)s;
    if (kin == 0) {
        float sb = 0.0f;
#pragma unroll
        for (int d = 0; d < 16; ++d)
            sb = fmaf(b[h * 16 + d], rel[(h * 16 + d) * 16 + e], sb);
        beff[n] = sb;
    }
}

// ---------------------------------------------------------------------------
// WMMA GEMM: C[M x 128] = epilogue( (act_in? gelu(A) : A)[M x K] @ W + bias )
// Wt is f16, transposed/N-major: Wt[n*K + k] == W[k][n]
// W-chunk staging goes through the Tensor Data Mover (tensor_load_to_lds) with
// TDM LDS padding reproducing the 40-half padded row stride; falls back to a
// cooperative copy when the builtin is unavailable.
// block = 256 threads (8 waves), tile = 128 rows x 128 cols
// ---------------------------------------------------------------------------
__launch_bounds__(256)
__global__ void hgt_gemm_wmma(const float* __restrict__ A, const _Float16* __restrict__ Wt,
                              const float* __restrict__ bias, float* __restrict__ C,
                              int M, int K, int act_in, int act_out,
                              const float* __restrict__ skip_h,
                              const float* __restrict__ skip_s) {
    __shared__ _Float16 sA[128 * 40];   // 128 rows x 32 k (+8 pad)
    __shared__ _Float16 sW[128 * 40];   // 128 n   x 32 k (+8 pad)

    const int tid    = threadIdx.x;
    const int wave   = tid >> 5;
    const int lane   = tid & 31;
    const int lane15 = lane & 15;
    const int hsel   = lane >> 4;
    const int row0   = blockIdx.x * 128;

    const int crow = tid >> 1;   // cooperative-load row 0..127
    const int cseg = tid & 1;    // which 16-k half-chunk

#ifdef HGT_HAS_TDM
    // LDS byte offset of sW: low 32 bits of the generic (LDS-aperture) address
    const unsigned sw_lds_base = (unsigned)(size_t)(&sW[0]);
#endif

    v8f acc[8] = {};

    for (int k0 = 0; k0 < K; k0 += 32) {
        __syncthreads();
        // ---- stage A chunk (f32 -> f16, optional gelu) ----
        {
            _Float16 t16[16];
            int grow = row0 + crow;
            if (grow < M) {
                const float4* ap = (const float4*)(A + (size_t)grow * K + k0 + cseg * 16);
                if (k0 + 32 < K) __builtin_prefetch((const void*)(ap + 8), 0, 1);
#pragma unroll
                for (int j = 0; j < 4; ++j) {
                    float4 f = ap[j];
                    float x0 = f.x, x1 = f.y, x2 = f.z, x3 = f.w;
                    if (act_in == 1) {
                        x0 = gelu_exact(x0); x1 = gelu_exact(x1);
                        x2 = gelu_exact(x2); x3 = gelu_exact(x3);
                    }
                    t16[j * 4 + 0] = (_Float16)x0;
                    t16[j * 4 + 1] = (_Float16)x1;
                    t16[j * 4 + 2] = (_Float16)x2;
                    t16[j * 4 + 3] = (_Float16)x3;
                }
            } else {
#pragma unroll
                for (int j = 0; j < 16; ++j) t16[j] = (_Float16)0.0f;
            }
            v8h* dst = (v8h*)&sA[crow * 40 + cseg * 16];
            dst[0] = *(v8h*)&t16[0];
            dst[1] = *(v8h*)&t16[8];
        }

        // ---- stage W chunk: 128 x 32 f16 tile ----
#ifdef HGT_HAS_TDM
        if (tid < 32) {   // one wave issues the TDM transfer (EXEC ignored by TDM)
            const unsigned long long wga =
                (unsigned long long)(size_t)Wt + (unsigned long long)k0 * 2ull;
            // D# group 0: count=1 | lds_addr | global_addr(57b) | type=2
            u32x4 g0 = { 1u,
                         sw_lds_base,
                         (unsigned)(wga & 0xFFFFFFFFull),
                         (unsigned)((wga >> 32) & 0x01FFFFFFull) | 0x80000000u };
            // D# group 1:
            //  dw0: data_size=2B (code1)<<16 | pad_enable<<20 | pad_interval=3(16dw)<<22
            //       | pad_amount=3(4dw)<<25   -> row stride 16+4 dwords = 80B = 40 halves
            //  dw1[31:16]=tensor_dim0.lo16(K)   dw2[31:16]=tensor_dim1.lo16(128)
            //  dw3[31:16]=tile_dim0(32)         dw4[15:0]=tile_dim1(128), tile_dim2=0
            //  dw5=tensor_dim0_stride.lo32(K)
            i32x8 g1 = { (int)0x06D10000u,
                         (int)((unsigned)(K & 0xFFFF) << 16),
                         (int)(128u << 16),
                         (int)(32u << 16),
                         128, K, 0, 0 };
            i32x4 gz = { 0, 0, 0, 0 };
            i32x8 g4z = { 0, 0, 0, 0, 0, 0, 0, 0 };
            // 6-arg form (clang-23 / amdgpu-toolchain): extra int32x8 group + cpol
            __builtin_amdgcn_tensor_load_to_lds(g0, g1, gz, gz, g4z, 0);
            __builtin_amdgcn_s_wait_tensorcnt(0);
        }
#else
        {
            const _Float16* wp = Wt + (size_t)crow * K + k0 + cseg * 16;
            v8h w0 = *(const v8h*)wp;
            v8h w1 = *(const v8h*)(wp + 8);
            v8h* dst = (v8h*)&sW[crow * 40 + cseg * 16];
            dst[0] = w0;
            dst[1] = w1;
        }
#endif
        __syncthreads();

        // ---- fragments + 8 WMMAs per wave ----
        const int arow = wave * 16 + lane15;
        v8h alo = *(const v8h*)&sA[arow * 40 + hsel * 8];
        v8h ahi = *(const v8h*)&sA[arow * 40 + 16 + hsel * 8];
        v16h af = CAT16(alo, ahi);
#pragma unroll
        for (int c = 0; c < 8; ++c) {
            const int brow = c * 16 + lane15;
            v8h blo = *(const v8h*)&sW[brow * 40 + hsel * 8];
            v8h bhi = *(const v8h*)&sW[brow * 40 + 16 + hsel * 8];
            v16h bf = CAT16(blo, bhi);
            acc[c] = __builtin_amdgcn_wmma_f32_16x16x32_f16(
                false, af, false, bf, (short)0, acc[c], false, false);
        }
    }

    // ---- epilogue: bias, relu, gated skip ----
    float gate = 1.0f;
    if (skip_s) gate = 1.0f / (1.0f + __expf(-*skip_s));
#pragma unroll
    for (int c = 0; c < 8; ++c) {
        int col = c * 16 + lane15;
        float bcol = bias ? bias[col] : 0.0f;
#pragma unroll
        for (int r = 0; r < 8; ++r) {
            int row = row0 + wave * 16 + hsel * 8 + r;
            if (row < M) {
                float v = acc[c][r] + bcol;
                if (act_out == 1) v = v > 0.0f ? v : 0.0f;
                if (skip_h) v = gate * v + (1.0f - gate) * skip_h[(size_t)row * 128 + col];
                C[(size_t)row * 128 + col] = v;
            }
        }
    }
}

// ---------------------------------------------------------------------------
// edge kernels
// ---------------------------------------------------------------------------
__global__ void hgt_edge_scores(const int* __restrict__ ei, int E,
                                const float* __restrict__ qdst, const float* __restrict__ ksrc,
                                const float* __restrict__ p_rel, float inv_sqrt_d,
                                float* __restrict__ alpha, unsigned* __restrict__ segmax) {
    int e = blockIdx.x * blockDim.x + threadIdx.x;
    if (e >= E) return;
    int si = ei[e];
    int di = ei[E + e];
    const float4* qp = (const float4*)(qdst + (size_t)di * 128);
    const float4* kp = (const float4*)(ksrc + (size_t)si * 128);
#pragma unroll
    for (int h = 0; h < 8; ++h) {
        float s = 0.0f;
#pragma unroll
        for (int j = 0; j < 4; ++j) {
            float4 a = qp[h * 4 + j];
            float4 b = kp[h * 4 + j];
            s = fmaf(a.x, b.x, s); s = fmaf(a.y, b.y, s);
            s = fmaf(a.z, b.z, s); s = fmaf(a.w, b.w, s);
        }
        s *= p_rel[h] * inv_sqrt_d;
        alpha[(size_t)e * 8 + h] = s;
        atomicMax(&segmax[(size_t)di * 8 + h], enc_ord(s));
    }
}

__global__ void hgt_edge_exp(const int* __restrict__ ei, int E,
                             float* __restrict__ alpha, const unsigned* __restrict__ segmax,
                             float* __restrict__ segsum) {
    int e = blockIdx.x * blockDim.x + threadIdx.x;
    if (e >= E) return;
    int di = ei[E + e];
#pragma unroll
    for (int h = 0; h < 8; ++h) {
        float m = dec_ord(segmax[(size_t)di * 8 + h]);
        if (!(fabsf(m) < 1.0e38f)) m = 0.0f;   // matches ref: non-finite -> 0
        float ex = __expf(alpha[(size_t)e * 8 + h] - m);
        alpha[(size_t)e * 8 + h] = ex;
        atomicAdd(&segsum[(size_t)di * 8 + h], ex);
    }
}

__global__ void hgt_edge_message(const int* __restrict__ ei, int E,
                                 const float* __restrict__ vrel, const float* __restrict__ alpha,
                                 const float* __restrict__ segsum, float* __restrict__ out) {
    int e = blockIdx.x * blockDim.x + threadIdx.x;
    if (e >= E) return;
    int si = ei[e];
    int di = ei[E + e];
    float w[8];
#pragma unroll
    for (int h = 0; h < 8; ++h)
        w[h] = alpha[(size_t)e * 8 + h] / (segsum[(size_t)di * 8 + h] + 1e-16f);
    const float4* vp = (const float4*)(vrel + (size_t)si * 128);
    float* op = out + (size_t)di * 128;
#pragma unroll 4
    for (int c4 = 0; c4 < 32; ++c4) {
        float4 v = vp[c4];
        float ww = w[c4 >> 2];
        atomicAdd(op + c4 * 4 + 0, v.x * ww);
        atomicAdd(op + c4 * 4 + 1, v.y * ww);
        atomicAdd(op + c4 * 4 + 2, v.z * ww);
        atomicAdd(op + c4 * 4 + 3, v.w * ww);
    }
}

// ---------------------------------------------------------------------------
// final projection: [M x 128] @ [128 x 2] + b
// ---------------------------------------------------------------------------
__global__ void hgt_lin_out(const float* __restrict__ h, const float* __restrict__ W,
                            const float* __restrict__ b, float* __restrict__ out, int M) {
    int i = blockIdx.x * blockDim.x + threadIdx.x;
    if (i >= M) return;
    const float* hp = h + (size_t)i * 128;
    float o0 = b[0], o1 = b[1];
#pragma unroll 8
    for (int c = 0; c < 128; ++c) {
        float x = hp[c];
        o0 = fmaf(x, W[c * 2 + 0], o0);
        o1 = fmaf(x, W[c * 2 + 1], o1);
    }
    out[(size_t)i * 2 + 0] = o0;
    out[(size_t)i * 2 + 1] = o1;
}

// ---------------------------------------------------------------------------
// host orchestration
// ---------------------------------------------------------------------------
extern "C" void kernel_launch(void* const* d_in, const int* in_sizes, int n_in,
                              void* d_out, int out_size, void* d_ws, size_t ws_size,
                              hipStream_t stream) {
    (void)n_in; (void)out_size; (void)ws_size;
    const int Ni = in_sizes[0] / 256;     // items
    const int Nu = in_sizes[1] / 256;     // users
    const int Eu2i = in_sizes[2] / 2;
    const int Ei2u = in_sizes[3] / 2;

    const float* x_item = (const float*)d_in[0];
    const float* x_user = (const float*)d_in[1];
    const int*   ei_u2i = (const int*)d_in[2];
    const int*   ei_i2u = (const int*)d_in[3];
    auto P = [&](int i) { return (const float*)d_in[i]; };
    // params leaf order (JAX sorted keys): layers -> lin_in -> lin_out
    // per layer: a(i.W,i.b,u.W,u.b) a_rel(i2u,u2i) k(4) m_rel(i2u,u2i)
    //            p_rel(i2u,u2i) q(4) skip(i,u) v(4)
    const int LB = 4, LS = 24;
    const int LIN_IN = 4 + 2 * LS;        // 52: item.W, item.b, user.W, user.b
    const int LIN_OUT = LIN_IN + 4;       // 56: W, b

    char* ws = (char*)d_ws;
    size_t off = 0;
    auto alloc = [&](size_t bytes) -> void* {
        void* p = ws + off;
        off = (off + bytes + 255) & ~(size_t)255;
        return p;
    };
    const size_t NBi = (size_t)Ni * 128 * sizeof(float);
    const size_t NBu = (size_t)Nu * 128 * sizeof(float);
    const size_t NBmax = NBi > NBu ? NBi : NBu;
    const int Nmax = Ni > Nu ? Ni : Nu;
    const int Emax = Eu2i > Ei2u ? Eu2i : Ei2u;

    float* h_i  = (float*)alloc(NBi);
    float* h_u  = (float*)alloc(NBu);
    float* nh_i = (float*)alloc(NBi);
    float* nh_u = (float*)alloc(NBu);
    float* q_i  = (float*)alloc(NBi);
    float* q_u  = (float*)alloc(NBu);
    float* krel = (float*)alloc(NBmax);
    float* vrel = (float*)alloc(NBmax);
    float* out_i = (float*)alloc(NBi);
    float* out_u = (float*)alloc(NBu);
    float* alpha = (float*)alloc((size_t)Emax * 8 * sizeof(float));
    unsigned* segmax = (unsigned*)alloc((size_t)Nmax * 8 * sizeof(unsigned));
    float* segsum = (float*)alloc((size_t)Nmax * 8 * sizeof(float));
    _Float16* w_lin_i = (_Float16*)alloc((size_t)256 * 128 * 2);
    _Float16* w_lin_u = (_Float16*)alloc((size_t)256 * 128 * 2);

    const dim3 B256(256);
    auto gemm = [&](const float* A, const _Float16* Wt, const float* bias, float* C,
                    int M, int K, int ai, int ao, const float* sh, const float* ss) {
        hgt_gemm_wmma<<<dim3((M + 127) / 128), B256, 0, stream>>>(A, Wt, bias, C, M, K, ai, ao, sh, ss);
    };
    auto fillf = [&](float* p, float v, long n) {
        hgt_fill_f32<<<dim3((unsigned)((n + 255) / 256)), B256, 0, stream>>>(p, v, n);
    };
    auto fillu = [&](unsigned* p, unsigned v, long n) {
        hgt_fill_u32<<<dim3((unsigned)((n + 255) / 256)), B256, 0, stream>>>(p, v, n);
    };

    // ---- lin_in: h = relu(x @ W + b) ----
    hgt_prep_plain<<<dim3(256 * 128 / 256), B256, 0, stream>>>(P(LIN_IN + 0), 256, w_lin_i);
    hgt_prep_plain<<<dim3(256 * 128 / 256), B256, 0, stream>>>(P(LIN_IN + 2), 256, w_lin_u);
    gemm(x_item, w_lin_i, P(LIN_IN + 1), h_i, Ni, 256, 0, 1, nullptr, nullptr);
    gemm(x_user, w_lin_u, P(LIN_IN + 3), h_u, Nu, 256, 0, 1, nullptr, nullptr);

    // ---- HGT layers ----
    for (int l = 0; l < 2; ++l) {
        const int L = LB + LS * l;
        _Float16* wq_i = (_Float16*)alloc(128 * 128 * 2);
        _Float16* wq_u = (_Float16*)alloc(128 * 128 * 2);
        _Float16* wk_u2i = (_Float16*)alloc(128 * 128 * 2);
        _Float16* wv_u2i = (_Float16*)alloc(128 * 128 * 2);
        _Float16* wk_i2u = (_Float16*)alloc(128 * 128 * 2);
        _Float16* wv_i2u = (_Float16*)alloc(128 * 128 * 2);
        _Float16* wa_i = (_Float16*)alloc(128 * 128 * 2);
        _Float16* wa_u = (_Float16*)alloc(128 * 128 * 2);
        float* bk_u2i = (float*)alloc(128 * 4);
        float* bv_u2i = (float*)alloc(128 * 4);
        float* bk_i2u = (float*)alloc(128 * 4);
        float* bv_i2u = (float*)alloc(128 * 4);

        const dim3 G64(64), G128b(128 * 128 / 256);
        hgt_prep_plain<<<G128b, B256, 0, stream>>>(P(L + 14), 128, wq_i);  // q.item.W
        hgt_prep_plain<<<G128b, B256, 0, stream>>>(P(L + 16), 128, wq_u);  // q.user.W
        hgt_prep_plain<<<G128b, B256, 0, stream>>>(P(L + 0), 128, wa_i);   // a.item.W
        hgt_prep_plain<<<G128b, B256, 0, stream>>>(P(L + 2), 128, wa_u);   // a.user.W
        // u2i: src=user -> fold a_rel.u2i into k.user, m_rel.u2i into v.user
        hgt_prep_rel<<<G64, B256, 0, stream>>>(P(L + 8), P(L + 9), P(L + 5), wk_u2i, bk_u2i);
        hgt_prep_rel<<<G64, B256, 0, stream>>>(P(L + 22), P(L + 23), P(L + 11), wv_u2i, bv_u2i);
        // i2u: src=item -> a_rel.i2u into k.item, m_rel.i2u into v.item
        hgt_prep_rel<<<G64, B256, 0, stream>>>(P(L + 6), P(L + 7), P(L + 4), wk_i2u, bk_i2u);
        hgt_prep_rel<<<G64, B256, 0, stream>>>(P(L + 20), P(L + 21), P(L + 10), wv_i2u, bv_i2u);

        // q projections (destinations)
        gemm(h_i, wq_i, P(L + 15), q_i, Ni, 128, 0, 0, nullptr, nullptr);
        gemm(h_u, wq_u, P(L + 17), q_u, Nu, 128, 0, 0, nullptr, nullptr);

        // ---- relation u2i (src=user, dst=item) ----
        gemm(h_u, wk_u2i, bk_u2i, krel, Nu, 128, 0, 0, nullptr, nullptr);
        gemm(h_u, wv_u2i, bv_u2i, vrel, Nu, 128, 0, 0, nullptr, nullptr);
        fillu(segmax, ENC_NEG_INF, (long)Ni * 8);
        fillf(segsum, 0.0f, (long)Ni * 8);
        fillf(out_i, 0.0f, (long)Ni * 128);
        {
            dim3 GE((Eu2i + 255) / 256);
            hgt_edge_scores<<<GE, B256, 0, stream>>>(ei_u2i, Eu2i, q_i, krel, P(L + 13), 0.25f, alpha, segmax);
            hgt_edge_exp<<<GE, B256, 0, stream>>>(ei_u2i, Eu2i, alpha, segmax, segsum);
            hgt_edge_message<<<GE, B256, 0, stream>>>(ei_u2i, Eu2i, vrel, alpha, segsum, out_i);
        }

        // ---- relation i2u (src=item, dst=user) ----
        gemm(h_i, wk_i2u, bk_i2u, krel, Ni, 128, 0, 0, nullptr, nullptr);
        gemm(h_i, wv_i2u, bv_i2u, vrel, Ni, 128, 0, 0, nullptr, nullptr);
        fillu(segmax, ENC_NEG_INF, (long)Nu * 8);
        fillf(segsum, 0.0f, (long)Nu * 8);
        fillf(out_u, 0.0f, (long)Nu * 128);
        {
            dim3 GE((Ei2u + 255) / 256);
            hgt_edge_scores<<<GE, B256, 0, stream>>>(ei_i2u, Ei2u, q_u, krel, P(L + 12), 0.25f, alpha, segmax);
            hgt_edge_exp<<<GE, B256, 0, stream>>>(ei_i2u, Ei2u, alpha, segmax, segsum);
            hgt_edge_message<<<GE, B256, 0, stream>>>(ei_i2u, Ei2u, vrel, alpha, segsum, out_u);
        }

        // ---- a-linear with fused gelu-on-input + gated skip ----
        gemm(out_i, wa_i, P(L + 1), nh_i, Ni, 128, 1, 0, h_i, P(L + 18));
        gemm(out_u, wa_u, P(L + 3), nh_u, Nu, 128, 1, 0, h_u, P(L + 19));

        float* t;
        t = h_i; h_i = nh_i; nh_i = t;
        t = h_u; h_u = nh_u; nh_u = t;
    }

    // ---- lin_out on items ----
    hgt_lin_out<<<dim3((Ni + 255) / 256), B256, 0, stream>>>(h_i, P(LIN_OUT), P(LIN_OUT + 1), (float*)d_out, Ni);
}
